// AttnBlock_79517024518472
// MI455X (gfx1250) — compile-verified, matrix operand order checked by reading, not run on hardware
//
#include <hip/hip_runtime.h>
#include <math.h>

// MI455X / gfx1250: wave32, WMMA 16x16x32 f16->f32.
typedef __attribute__((ext_vector_type(16))) _Float16 v16h;
typedef __attribute__((ext_vector_type(8)))  float    v8f;

#define DD   128   // feature dim
#define NHD  8     // heads
#define TDIM 512   // time-embedding dim
#define WPFRAGS (8 * 4 * 32)         // col-tiles * k-steps * lanes
#define WPHALF  (WPFRAGS * 16)       // halves per packed 128x128 weight

// ---------------- helpers ----------------

__device__ __forceinline__ float silu_(float x) {
    return x / (1.0f + __expf(-x));
}

// reduce across the 4 lanes of an aligned lane-quad (one 16-channel group)
__device__ __forceinline__ float quadSum_(float v) {
    v += __shfl_xor(v, 1, 32);
    v += __shfl_xor(v, 2, 32);
    return v;
}

// float atomic max via sign-split int/uint atomics (init value must be -inf)
__device__ __forceinline__ void atomicMaxFloat_(float* addr, float val) {
    if (val >= 0.0f)
        atomicMax(reinterpret_cast<int*>(addr), __float_as_int(val));
    else
        atomicMin(reinterpret_cast<unsigned int*>(addr), __float_as_uint(val));
}

// ---------------- fill ----------------

__global__ void fill_f32(float* __restrict__ p, float v, size_t n) {
    size_t i = blockIdx.x * (size_t)blockDim.x + threadIdx.x;
    size_t stride = (size_t)gridDim.x * blockDim.x;
    for (; i < n; i += stride) p[i] = v;
}

// ---------------- pack W[128][128] f32 -> f16 WMMA B-fragment order ----------
// fragment f = (tile*4 + ks)*32 + lane holds the 16 halves of the B operand
// for column tile `tile`, K-step `ks`, lane `lane`, in register order hh=0..15.

__global__ void pack_w_f16(const float* __restrict__ W, _Float16* __restrict__ Wp) {
    int idx = blockIdx.x * blockDim.x + threadIdx.x;   // 0 .. WPHALF-1
    if (idx >= WPHALF) return;
    const int hh     = idx & 15;
    const int lane   = (idx >> 4) & 31;
    const int ks     = (idx >> 9) & 3;
    const int tile   = idx >> 11;
    const int laneHi = lane >> 4;
    const int vv = hh >> 1, pos = hh & 1;
    const int kk = ((vv < 4) ? (2 * vv + pos) : (2 * vv + 8 + pos)) + laneHi * 8;
    const int k   = ks * 32 + kk;
    const int col = tile * 16 + (lane & 15);
    Wp[idx] = (_Float16)W[k * DD + col];
}

// ---------------- time MLP: tnode = silu(t_emb) @ tm_w + tm_b ----------------

__global__ void time_mlp(const float* __restrict__ t_emb,
                         const float* __restrict__ tm_w,
                         const float* __restrict__ tm_b,
                         float* __restrict__ tnode) {
    int c = threadIdx.x;            // 128 threads
    float acc = tm_b[c];
    for (int j = 0; j < TDIM; ++j) {
        float t = t_emb[j];
        acc += silu_(t) * tm_w[j * DD + c];
    }
    tnode[c] = acc;
}

// ---------------- Y[n,128] = X[n,128] @ W[128,128] (+bias) via WMMA ----------
// 256 threads = 8 waves; block owns 16 rows; wave w owns cols 16w..16w+15.
// B comes pre-packed in fragment order: one 32B vector load per K-step.

__global__ __launch_bounds__(256)
void gemm128_wmma(const float* __restrict__ X, const _Float16* __restrict__ Wp,
                  const float* __restrict__ bias, float* __restrict__ Y, int n) {
    __shared__ _Float16 sA[16 * DD];
    const int tid  = threadIdx.x;
    const int row0 = blockIdx.x * 16;

    // stage 16x128 A tile (f32 -> f16) into LDS, shared by all 8 waves
    for (int i = tid; i < 16 * DD; i += 256) {
        int m = i >> 7, k = i & 127;
        int r = row0 + m; if (r >= n) r = n - 1;
        sA[i] = (_Float16)X[(size_t)r * DD + k];
    }
    __syncthreads();

    const int wave   = tid >> 5;
    const int lane   = tid & 31;
    const int laneHi = lane >> 4;      // 0: k 0-7,16-23; 1: k 8-15,24-31 (A/B)
    const int m16    = lane & 15;
    const int col    = wave * 16 + m16;
    const v16h* __restrict__ bfrag =
        reinterpret_cast<const v16h*>(Wp) + (size_t)(wave * 4) * 32 + lane;

    v8f acc = {};
#pragma unroll
    for (int ks = 0; ks < 4; ++ks) {
        // A fragment: two contiguous 8-half runs from LDS (2x ds_load_b128)
        v16h a;
        const int base = m16 * DD + ks * 32 + laneHi * 8;
#pragma unroll
        for (int j = 0; j < 8; ++j) {
            a[j]     = sA[base + j];
            a[j + 8] = sA[base + 16 + j];
        }
        // B fragment: one contiguous 32B global load (2x global_load_b128)
        const v16h b = bfrag[ks * 32];
        acc = __builtin_amdgcn_wmma_f32_16x16x32_f16(
                  false, a, false, b, (short)0, acc, false, false);
    }

    const float bv = bias ? bias[col] : 0.0f;
    if (row0 + 16 <= n) {
        // full tile: unguarded, coalesced store clause
#pragma unroll
        for (int r = 0; r < 8; ++r)
            Y[(size_t)(row0 + r + laneHi * 8) * DD + col] = acc[r] + bv;
    } else {
#pragma unroll
        for (int r = 0; r < 8; ++r) {
            const int row = row0 + r + laneHi * 8;   // C/D: lanes>=16 hold M=8..15
            if (row < n) Y[(size_t)row * DD + col] = acc[r] + bv;
        }
    }
}

// ---------------- per-edge message + groupnorm + silu + scatter --------------
// one wave per edge; lane holds channels 4l..4l+3; group(16ch) = 4 lanes

__global__ __launch_bounds__(256)
void edge_conv(const float* __restrict__ xd, const float* __restrict__ xs,
               const float* __restrict__ ea, const int* __restrict__ src,
               const int* __restrict__ dst, const float* __restrict__ We,
               const float* __restrict__ gg, const float* __restrict__ bb,
               float* __restrict__ sAcc, float* __restrict__ cnt, int E) {
    const int wid = (int)((blockIdx.x * (size_t)blockDim.x + threadIdx.x) >> 5);
    if (wid >= E) return;
    const int lane = threadIdx.x & 31;
    const int c = lane * 4;
    const int d = dst[wid], s = src[wid];

    const float4 md = *(const float4*)(xd + (size_t)d * DD + c);
    const float4 ms = *(const float4*)(xs + (size_t)s * DD + c);
    const float4 e4 = *(const float4*)(ea + (size_t)wid * 4);
    const float4 w0 = *(const float4*)(We + 0 * DD + c);
    const float4 w1 = *(const float4*)(We + 1 * DD + c);
    const float4 w2 = *(const float4*)(We + 2 * DD + c);
    const float4 w3 = *(const float4*)(We + 3 * DD + c);

    float m0 = md.x + ms.x + e4.x * w0.x + e4.y * w1.x + e4.z * w2.x + e4.w * w3.x;
    float m1 = md.y + ms.y + e4.x * w0.y + e4.y * w1.y + e4.z * w2.y + e4.w * w3.y;
    float m2 = md.z + ms.z + e4.x * w0.z + e4.y * w1.z + e4.z * w2.z + e4.w * w3.z;
    float m3 = md.w + ms.w + e4.x * w0.w + e4.y * w1.w + e4.z * w2.w + e4.w * w3.w;

    // groupnorm over 16 channels (4 lanes), biased variance
    float mean = quadSum_(m0 + m1 + m2 + m3) * 0.0625f;
    float d0 = m0 - mean, d1 = m1 - mean, d2 = m2 - mean, d3 = m3 - mean;
    float var = quadSum_(d0 * d0 + d1 * d1 + d2 * d2 + d3 * d3) * 0.0625f;
    float inv = rsqrtf(var + 1e-5f);

    const float4 g4 = *(const float4*)(gg + c);
    const float4 b4 = *(const float4*)(bb + c);
    float y0 = silu_(d0 * inv * g4.x + b4.x);
    float y1 = silu_(d1 * inv * g4.y + b4.y);
    float y2 = silu_(d2 * inv * g4.z + b4.z);
    float y3 = silu_(d3 * inv * g4.w + b4.w);

    float* sp = sAcc + (size_t)d * DD + c;
    atomicAdd(sp + 0, y0); atomicAdd(sp + 1, y1);
    atomicAdd(sp + 2, y2); atomicAdd(sp + 3, y3);
    if (lane == 0) atomicAdd(cnt + d, 1.0f);
}

// ---------------- node epilogue: mean-div + groupnorm + silu (+tvec)(+res) ---

__global__ __launch_bounds__(256)
void node_post(const float* __restrict__ tmp, const float* __restrict__ cnt,
               const float* __restrict__ b2, const float* __restrict__ gg,
               const float* __restrict__ bb, const float* __restrict__ tvec,
               const float* __restrict__ res, float* __restrict__ out, int n) {
    const int wid = (int)((blockIdx.x * (size_t)blockDim.x + threadIdx.x) >> 5);
    if (wid >= n) return;
    const int lane = threadIdx.x & 31;
    const int c = lane * 4;

    const float cv = cnt[wid];
    const float ic = 1.0f / fmaxf(cv, 1.0f);
    const float4 t4  = *(const float4*)(tmp + (size_t)wid * DD + c);
    const float4 b24 = *(const float4*)(b2 + c);
    float m0 = (t4.x + cv * b24.x) * ic;
    float m1 = (t4.y + cv * b24.y) * ic;
    float m2 = (t4.z + cv * b24.z) * ic;
    float m3 = (t4.w + cv * b24.w) * ic;

    float mean = quadSum_(m0 + m1 + m2 + m3) * 0.0625f;
    float d0 = m0 - mean, d1 = m1 - mean, d2 = m2 - mean, d3 = m3 - mean;
    float var = quadSum_(d0 * d0 + d1 * d1 + d2 * d2 + d3 * d3) * 0.0625f;
    float inv = rsqrtf(var + 1e-5f);

    const float4 g4 = *(const float4*)(gg + c);
    const float4 b4 = *(const float4*)(bb + c);
    float y0 = silu_(d0 * inv * g4.x + b4.x);
    float y1 = silu_(d1 * inv * g4.y + b4.y);
    float y2 = silu_(d2 * inv * g4.z + b4.z);
    float y3 = silu_(d3 * inv * g4.w + b4.w);

    if (tvec) {
        const float4 tv = *(const float4*)(tvec + c);
        y0 += tv.x; y1 += tv.y; y2 += tv.z; y3 += tv.w;
    }
    if (res) {
        const float4 r4 = *(const float4*)(res + (size_t)wid * DD + c);
        y0 += r4.x; y1 += r4.y; y2 += r4.z; y3 += r4.w;
    }
    float4 o; o.x = y0; o.y = y1; o.z = y2; o.w = y3;
    *(float4*)(out + (size_t)wid * DD + c) = o;
}

// ---------------- attention: logits + segment max ----------------------------
// one wave per edge; head h = lane/4; each lane holds 4 of the 16 head dims

__global__ __launch_bounds__(256)
void edge_logits(const float* __restrict__ q, const float* __restrict__ k,
                 const float* __restrict__ ea, const int* __restrict__ src,
                 const int* __restrict__ dst, const float* __restrict__ ew,
                 const float* __restrict__ eb, float* __restrict__ logits,
                 float* __restrict__ amax, int E) {
    const int wid = (int)((blockIdx.x * (size_t)blockDim.x + threadIdx.x) >> 5);
    if (wid >= E) return;
    const int lane = threadIdx.x & 31;
    const int c = lane * 4;
    const int h = lane >> 2;
    const int d = dst[wid], s = src[wid];

    const float4 q4 = *(const float4*)(q + (size_t)d * DD + c);
    const float4 k4 = *(const float4*)(k + (size_t)s * DD + c);
    float p = q4.x * k4.x + q4.y * k4.y + q4.z * k4.z + q4.w * k4.w;
    p = quadSum_(p);   // full 16-dim dot per head

    if ((lane & 3) == 0) {
        const float4 e4 = *(const float4*)(ea + (size_t)wid * 4);
        float lg = p * 0.25f   // HD^-0.5 = 1/4
                 + e4.x * ew[0 * NHD + h] + e4.y * ew[1 * NHD + h]
                 + e4.z * ew[2 * NHD + h] + e4.w * ew[3 * NHD + h] + eb[h];
        logits[(size_t)wid * NHD + h] = lg;
        atomicMaxFloat_(amax + (size_t)d * NHD + h, lg);
    }
}

// ---------------- attention: exp + segment sum (thread per edge*head) --------

__global__ void edge_softmax(float* __restrict__ logits,
                             const float* __restrict__ amax,
                             float* __restrict__ ssum,
                             const int* __restrict__ dst, int E) {
    size_t idx = blockIdx.x * (size_t)blockDim.x + threadIdx.x;
    if (idx >= (size_t)E * NHD) return;
    int e = (int)(idx >> 3), h = (int)(idx & 7);
    int d = dst[e];
    float pe = __expf(logits[idx] - amax[(size_t)d * NHD + h]);
    logits[idx] = pe;
    atomicAdd(ssum + (size_t)d * NHD + h, pe);
}

// ---------------- attention: weighted V scatter ------------------------------

__global__ __launch_bounds__(256)
void edge_accum(const float* __restrict__ logits, const float* __restrict__ ssum,
                const float* __restrict__ v, const int* __restrict__ src,
                const int* __restrict__ dst, float* __restrict__ o, int E) {
    const int wid = (int)((blockIdx.x * (size_t)blockDim.x + threadIdx.x) >> 5);
    if (wid >= E) return;
    const int lane = threadIdx.x & 31;
    const int c = lane * 4;
    const int h = lane >> 2;
    const int d = dst[wid], s = src[wid];

    float w = logits[(size_t)wid * NHD + h] /
              fmaxf(ssum[(size_t)d * NHD + h], 1e-6f);
    const float4 v4 = *(const float4*)(v + (size_t)s * DD + c);
    float* op = o + (size_t)d * DD + c;
    atomicAdd(op + 0, w * v4.x); atomicAdd(op + 1, w * v4.y);
    atomicAdd(op + 2, w * v4.z); atomicAdd(op + 3, w * v4.w);
}

// ---------------- final: out = res + groupnorm(o2) ---------------------------

__global__ __launch_bounds__(256)
void node_gn_add(const float* __restrict__ o2, const float* __restrict__ gg,
                 const float* __restrict__ bb, const float* __restrict__ res,
                 float* __restrict__ out, int n) {
    const int wid = (int)((blockIdx.x * (size_t)blockDim.x + threadIdx.x) >> 5);
    if (wid >= n) return;
    const int lane = threadIdx.x & 31;
    const int c = lane * 4;

    const float4 t4 = *(const float4*)(o2 + (size_t)wid * DD + c);
    float m0 = t4.x, m1 = t4.y, m2 = t4.z, m3 = t4.w;
    float mean = quadSum_(m0 + m1 + m2 + m3) * 0.0625f;
    float d0 = m0 - mean, d1 = m1 - mean, d2 = m2 - mean, d3 = m3 - mean;
    float var = quadSum_(d0 * d0 + d1 * d1 + d2 * d2 + d3 * d3) * 0.0625f;
    float inv = rsqrtf(var + 1e-5f);

    const float4 g4 = *(const float4*)(gg + c);
    const float4 b4 = *(const float4*)(bb + c);
    const float4 r4 = *(const float4*)(res + (size_t)wid * DD + c);
    float4 o;
    o.x = d0 * inv * g4.x + b4.x + r4.x;
    o.y = d1 * inv * g4.y + b4.y + r4.y;
    o.z = d2 * inv * g4.z + b4.z + r4.z;
    o.w = d3 * inv * g4.w + b4.w + r4.w;
    *(float4*)(out + (size_t)wid * DD + c) = o;
}

// ---------------- host orchestration ----------------------------------------

extern "C" void kernel_launch(void* const* d_in, const int* in_sizes, int n_in,
                              void* d_out, int out_size, void* d_ws, size_t ws_size,
                              hipStream_t stream) {
    (void)n_in; (void)out_size; (void)ws_size;
    const float* x     = (const float*)d_in[0];
    const int*   esrc  = (const int*)  d_in[1];
    const int*   edst  = (const int*)  d_in[2];
    const float* eattr = (const float*)d_in[3];
    const float* t_emb = (const float*)d_in[4];
    const float* c1_w1 = (const float*)d_in[5];
    const float* c1_b1 = (const float*)d_in[6];
    const float* c1_g  = (const float*)d_in[7];
    const float* c1_bt = (const float*)d_in[8];
    const float* c1_w2 = (const float*)d_in[9];
    const float* c1_b2 = (const float*)d_in[10];
    const float* c2_w1 = (const float*)d_in[11];
    const float* c2_b1 = (const float*)d_in[12];
    const float* c2_g  = (const float*)d_in[13];
    const float* c2_bt = (const float*)d_in[14];
    const float* c2_w2 = (const float*)d_in[15];
    const float* c2_b2 = (const float*)d_in[16];
    const float* tm_w  = (const float*)d_in[17];
    const float* tm_b  = (const float*)d_in[18];
    const float* n1_g  = (const float*)d_in[19];
    const float* n1_b  = (const float*)d_in[20];
    const float* n2_g  = (const float*)d_in[21];
    const float* n2_b  = (const float*)d_in[22];
    const float* qw    = (const float*)d_in[23];
    const float* qb    = (const float*)d_in[24];
    const float* kw    = (const float*)d_in[25];
    const float* kb    = (const float*)d_in[26];
    const float* vw    = (const float*)d_in[27];
    const float* vb    = (const float*)d_in[28];
    const float* ew    = (const float*)d_in[29];
    const float* eb    = (const float*)d_in[30];
    const float* ow    = (const float*)d_in[31];
    const float* ob    = (const float*)d_in[32];
    const float* an_g  = (const float*)d_in[33];
    const float* an_b  = (const float*)d_in[34];

    const int N = in_sizes[0] / DD;
    const int E = in_sizes[1];
    const size_t ND = (size_t)N * DD;

    float* ws  = (float*)d_ws;
    float* A   = ws;                       // xd / q
    float* B   = A + ND;                   // xs / k
    float* S   = B + ND;                   // segment accumulator / attn o
    float* CNT = S + ND;                   // [N]
    float* Hh  = CNT + N;                  // node features h / hfin
    float* T   = Hh + ND;                  // gemm tmp / v / o2
    float* LG  = T + ND;                   // logits [E, 8]
    float* AM  = LG + (size_t)E * NHD;     // [N, 8]
    float* SS  = AM + (size_t)N * NHD;     // [N, 8]
    float* TN  = SS + (size_t)N * NHD;     // [128] time vector
    // packed f16 weights (32B aligned only if N % 8 == 0: CNT holds N floats)
    _Float16* WP = (_Float16*)(TN + DD);
    _Float16* p_c1d = WP + 0 * WPHALF;
    _Float16* p_c1s = WP + 1 * WPHALF;
    _Float16* p_c1w2= WP + 2 * WPHALF;
    _Float16* p_c2d = WP + 3 * WPHALF;
    _Float16* p_c2s = WP + 4 * WPHALF;
    _Float16* p_c2w2= WP + 5 * WPHALF;
    _Float16* p_qw  = WP + 6 * WPHALF;
    _Float16* p_kw  = WP + 7 * WPHALF;
    _Float16* p_vw  = WP + 8 * WPHALF;
    _Float16* p_ow  = WP + 9 * WPHALF;

    const dim3 blk(256);
    const int gemmGrid = (N + 15) / 16;
    const int edgeGrid = (E + 7) / 8;      // 8 waves/block, wave per edge
    const int nodeGrid = (N + 7) / 8;
    const int ehGrid   = (int)(((size_t)E * NHD + 255) / 256);
    const int packGrid = (WPHALF + 255) / 256;

    // ---- one-time weight packing into WMMA fragment order (f16)
    pack_w_f16<<<packGrid, blk, 0, stream>>>(c1_w1,            p_c1d);
    pack_w_f16<<<packGrid, blk, 0, stream>>>(c1_w1 + 128 * DD, p_c1s);
    pack_w_f16<<<packGrid, blk, 0, stream>>>(c1_w2,            p_c1w2);
    pack_w_f16<<<packGrid, blk, 0, stream>>>(c2_w1,            p_c2d);
    pack_w_f16<<<packGrid, blk, 0, stream>>>(c2_w1 + 128 * DD, p_c2s);
    pack_w_f16<<<packGrid, blk, 0, stream>>>(c2_w2,            p_c2w2);
    pack_w_f16<<<packGrid, blk, 0, stream>>>(qw,               p_qw);
    pack_w_f16<<<packGrid, blk, 0, stream>>>(kw,               p_kw);
    pack_w_f16<<<packGrid, blk, 0, stream>>>(vw,               p_vw);
    pack_w_f16<<<packGrid, blk, 0, stream>>>(ow,               p_ow);

    time_mlp<<<1, 128, 0, stream>>>(t_emb, tm_w, tm_b, TN);

    // ---- conv1: b1 folded into xd projection; W2 pulled through segment_sum
    gemm128_wmma<<<gemmGrid, blk, 0, stream>>>(x, p_c1d, c1_b1,  A, N);
    gemm128_wmma<<<gemmGrid, blk, 0, stream>>>(x, p_c1s, nullptr, B, N);
    fill_f32<<<4096, 256, 0, stream>>>(S, 0.0f, ND);
    fill_f32<<<256,  256, 0, stream>>>(CNT, 0.0f, (size_t)N);
    edge_conv<<<edgeGrid, blk, 0, stream>>>(A, B, eattr, esrc, edst,
                                            c1_w1 + 256 * DD, c1_g, c1_bt, S, CNT, E);
    gemm128_wmma<<<gemmGrid, blk, 0, stream>>>(S, p_c1w2, nullptr, T, N);
    node_post<<<nodeGrid, blk, 0, stream>>>(T, CNT, c1_b2, n1_g, n1_b, TN, nullptr, Hh, N);

    // ---- conv2 (+ residual x)
    gemm128_wmma<<<gemmGrid, blk, 0, stream>>>(Hh, p_c2d, c2_b1,  A, N);
    gemm128_wmma<<<gemmGrid, blk, 0, stream>>>(Hh, p_c2s, nullptr, B, N);
    fill_f32<<<4096, 256, 0, stream>>>(S, 0.0f, ND);
    fill_f32<<<256,  256, 0, stream>>>(CNT, 0.0f, (size_t)N);
    edge_conv<<<edgeGrid, blk, 0, stream>>>(A, B, eattr, esrc, edst,
                                            c2_w1 + 256 * DD, c2_g, c2_bt, S, CNT, E);
    gemm128_wmma<<<gemmGrid, blk, 0, stream>>>(S, p_c2w2, nullptr, T, N);
    node_post<<<nodeGrid, blk, 0, stream>>>(T, CNT, c2_b2, n2_g, n2_b, nullptr, x, Hh, N);

    // ---- graph attention
    gemm128_wmma<<<gemmGrid, blk, 0, stream>>>(Hh, p_qw, qb, A, N);
    gemm128_wmma<<<gemmGrid, blk, 0, stream>>>(Hh, p_kw, kb, B, N);
    gemm128_wmma<<<gemmGrid, blk, 0, stream>>>(Hh, p_vw, vb, T, N);
    fill_f32<<<4096, 256, 0, stream>>>(S, 0.0f, ND);
    fill_f32<<<256,  256, 0, stream>>>(AM, -INFINITY, (size_t)N * NHD);
    fill_f32<<<256,  256, 0, stream>>>(SS, 0.0f,      (size_t)N * NHD);
    edge_logits<<<edgeGrid, blk, 0, stream>>>(A, B, eattr, esrc, edst, ew, eb, LG, AM, E);
    edge_softmax<<<ehGrid, blk, 0, stream>>>(LG, AM, SS, edst, E);
    edge_accum<<<edgeGrid, blk, 0, stream>>>(LG, SS, T, esrc, edst, S, E);
    gemm128_wmma<<<gemmGrid, blk, 0, stream>>>(S, p_ow, ob, T, N);   // o2 (v dead)
    node_gn_add<<<nodeGrid, blk, 0, stream>>>(T, an_g, an_b, Hh, (float*)d_out, N);
}
